// RGCNModel_24292335026208
// MI455X (gfx1250) — matrix operand order repaired and checked by reading, not compiled, not run on hardware
//
#include <hip/hip_runtime.h>
#include <hip/hip_bf16.h>

typedef __attribute__((ext_vector_type(16))) _Float16 v16h;
typedef __attribute__((ext_vector_type(8)))  _Float16 v8h;
typedef __attribute__((ext_vector_type(4)))  _Float16 v4h;
typedef __attribute__((ext_vector_type(8)))  float    v8f;
typedef __attribute__((ext_vector_type(4)))  unsigned int u32x4;
typedef __attribute__((ext_vector_type(8)))  int          i32x8;
typedef __attribute__((ext_vector_type(4)))  int          i32x4;

#define NN  100000   // nodes
#define NR  3        // relations
#define NE  200000   // edges per relation
#define FD  128      // feature dim (in = hid = out)
#define NQ  4096     // queries
#define SWP 136      // padded LDS row stride in halves (128 + 8 -> 272B, bank-conflict free)

#if __has_builtin(__builtin_amdgcn_tensor_load_to_lds)
#define USE_TDM 1
#else
#define USE_TDM 0
#endif

// ---------------- float -> half cast (vector-4) ----------------
__global__ void k_cast_half(const float* __restrict__ src, _Float16* __restrict__ dst, int n4) {
    int i = blockIdx.x * blockDim.x + threadIdx.x;
    if (i >= n4) return;
    float4 v = ((const float4*)src)[i];
    v4h o;
    o[0] = (_Float16)v.x; o[1] = (_Float16)v.y; o[2] = (_Float16)v.z; o[3] = (_Float16)v.w;
    ((v4h*)dst)[i] = o;
}

// ------------- W [r][k][n] f32 -> Wt [r][n][k] f16 (transpose) -------------
__global__ void k_prep_w(const float* __restrict__ W, _Float16* __restrict__ Wt, int total) {
    int i = blockIdx.x * blockDim.x + threadIdx.x;
    if (i >= total) return;
    int r   = i / (FD * FD);
    int rem = i - r * FD * FD;
    int n   = rem / FD;
    int k   = rem - n * FD;
    Wt[i] = (_Float16)W[((size_t)r * FD + k) * FD + n];
}

// ---------------- degree counting (exact: sums of 1.0f) ----------------
__global__ void k_deg(const int* __restrict__ es, const int* __restrict__ ed,
                      float* __restrict__ dsrc, float* __restrict__ ddst, int total) {
    int i = blockIdx.x * blockDim.x + threadIdx.x;
    if (i >= total) return;
    int r = i / NE;
    atomicAdd(&dsrc[(size_t)r * NN + es[i]], 1.0f);
    atomicAdd(&ddst[(size_t)r * NN + ed[i]], 1.0f);
}

// ---------------- deg -> rsqrt(clip(deg,1)) in place ----------------
__global__ void k_rsqrt(float* __restrict__ a, float* __restrict__ b, int total) {
    int i = blockIdx.x * blockDim.x + threadIdx.x;
    if (i >= total) return;
    float da = a[i]; a[i] = rsqrtf(da < 1.0f ? 1.0f : da);
    float db = b[i]; b[i] = rsqrtf(db < 1.0f ? 1.0f : db);
}

// ---------------- WMMA GEMM: Z[nrows,128] = Xh[nrows,128] @ W (via Wt[n][k]) ----
// One wave per 16-row band; 8 column tiles x 4 K-chunks of v_wmma_f32_16x16x32_f16.
// W tile staged into padded LDS by the Tensor Data Mover (pad = 16B per 256B row).
__global__ __launch_bounds__(256) void k_gemm_wmma(const _Float16* __restrict__ Xh,
                                                   const _Float16* __restrict__ Wt,
                                                   float* __restrict__ Z, int nrows) {
    __shared__ __align__(16) _Float16 sW[FD * SWP];   // 34 KB padded weight tile
    int tid = threadIdx.x;

#if USE_TDM
    if (tid < 32) {   // one wave issues the TDM descriptor (EXEC-independent DMA)
        unsigned           lds = (unsigned)(uintptr_t)(&sW[0]);
        unsigned long long ga  = (unsigned long long)(uintptr_t)Wt;
        u32x4 g0;
        g0[0] = 1u;                                            // count=1, user mode
        g0[1] = lds;                                           // lds_addr (bytes)
        g0[2] = (unsigned)(ga & 0xffffffffu);                  // global_addr[31:0]
        g0[3] = (unsigned)((ga >> 32) & 0x01ffffffu) | (2u << 30); // addr[56:32] | type=2
        i32x8 g1;
        g1[0] = (int)((1u << 16)       // data_size = 2 bytes
                    | (1u << 20)       // pad_enable
                    | (5u << 22)       // pad_interval: 64 DWORDs = 256B (one 128-half row)
                    | (3u << 25));     // pad_amount: 4 DWORDs = 16B
        g1[1] = (int)(128u << 16);     // tensor_dim0 = 128 (bits 79:48)
        g1[2] = (int)(128u << 16);     // tensor_dim1 = 128 (bits 111:80)
        g1[3] = (int)(128u << 16);     // tile_dim0  = 128 (bits 127:112)
        g1[4] = 128;                   // tile_dim1  = 128 (bits 143:128)
        g1[5] = 128;                   // tensor_dim0_stride = 128 (bits 191:160)
        g1[6] = 0;
        g1[7] = 0;
        i32x4 gz4 = {0, 0, 0, 0};
        i32x8 gz8 = {0, 0, 0, 0, 0, 0, 0, 0};
        __builtin_amdgcn_tensor_load_to_lds(g0, g1, gz4, gz4, gz8, 0);
        __builtin_amdgcn_s_wait_tensorcnt(0);
    }
#else
    {   // fallback: cooperative padded copy (row stride 17 int4 = 272B)
        const int4* g = (const int4*)Wt;
        int4*       s = (int4*)sW;
#pragma unroll
        for (int i = 0; i < 8; ++i) {
            int j = tid + i * 256;           // int4 index over 128x128 halves
            s[(j >> 4) * 17 + (j & 15)] = g[j];
        }
    }
#endif
    __syncthreads();

    int wave = tid >> 5;
    int lane = tid & 31;
    int row0 = (blockIdx.x * 8 + wave) << 4;
    if (row0 >= nrows) return;           // wave-uniform: EXEC stays all-ones for WMMA
    int rl = lane & 15;
    int hi = lane >> 4;

    // A layout (ISA 7.12.2, 16-bit A 16x32): lane<16 -> K {0..7,16..23}, lane>=16 -> K {8..15,24..31}
    const _Float16* xp = Xh + (size_t)(row0 + rl) * FD;
    v16h A[4];
#pragma unroll
    for (int kc = 0; kc < 4; ++kc) {
        v8h lo = *(const v8h*)(xp + kc * 32 + hi * 8);
        v8h hp = *(const v8h*)(xp + kc * 32 + 16 + hi * 8);
#pragma unroll
        for (int j = 0; j < 8; ++j) { A[kc][j] = lo[j]; A[kc][j + 8] = hp[j]; }
    }

#pragma unroll
    for (int ct = 0; ct < 8; ++ct) {
        int col0 = ct << 4;
        v8f acc = {};
#pragma unroll
        for (int kc = 0; kc < 4; ++kc) {
            // B layout: lane<16 -> N=rl, K = kc*32 + {0..7,16..23} ; lane>=16 -> K = kc*32 + {8..15,24..31}
            const _Float16* wp = sW + (size_t)(col0 + rl) * SWP + kc * 32 + hi * 8;
            v8h b0 = *(const v8h*)(wp);
            v8h b1 = *(const v8h*)(wp + 16);
            v16h b;
#pragma unroll
            for (int j = 0; j < 8; ++j) { b[j] = b0[j]; b[j + 8] = b1[j]; }
            acc = __builtin_amdgcn_wmma_f32_16x16x32_f16(false, A[kc], false, b,
                                                         (short)0, acc, false, false);
        }
        // D layout: element m -> row (row0 + hi*8 + m), col (col0 + rl)
        float* zp = Z + (size_t)(row0 + hi * 8) * FD + col0 + rl;
#pragma unroll
        for (int m = 0; m < 8; ++m) zp[(size_t)m * FD] = acc[m];
    }
}

// ------- per-edge: agg[ed] += Z[es] * s_src[es] * s_dst[ed] (one wave per edge) -------
__global__ void k_edge(const float* __restrict__ Z, const int* __restrict__ es,
                       const int* __restrict__ ed, const float* __restrict__ ssrc,
                       const float* __restrict__ sdst, float* __restrict__ agg, int nE) {
    int g    = blockIdx.x * blockDim.x + threadIdx.x;
    int e    = g >> 5;
    int lane = g & 31;
    if (e >= nE) return;
    int s = es[e], d = ed[e];
    float c = ssrc[s] * sdst[d];
    float4 z = ((const float4*)(Z + (size_t)s * FD))[lane];
    float* ap = agg + (size_t)d * FD + lane * 4;
    atomicAdd(ap + 0, z.x * c);
    atomicAdd(ap + 1, z.y * c);
    atomicAdd(ap + 2, z.z * c);
    atomicAdd(ap + 3, z.w * c);
}

// ------- finalize: t = tanh(agg + sum_r b[r][f]); optional f32 and/or f16 outputs -------
__global__ void k_finalize(const float* __restrict__ agg, const float* __restrict__ b,
                           float* __restrict__ outF, _Float16* __restrict__ outH, int total) {
    int i = blockIdx.x * blockDim.x + threadIdx.x;
    if (i >= total) return;
    int f = i & (FD - 1);
    float t = tanhf(agg[i] + b[f] + b[FD + f] + b[2 * FD + f]);
    if (outF) outF[i] = t;
    if (outH) outH[i] = (_Float16)t;
}

// ------- gather query rows: out[0:Q]=h[src], out[Q:2Q]=h[dst] -------
__global__ void k_gather(const float* __restrict__ h, const int* __restrict__ qs,
                         const int* __restrict__ qd, float* __restrict__ out, int Q) {
    int g    = blockIdx.x * blockDim.x + threadIdx.x;
    int q    = g >> 5;
    int lane = g & 31;
    if (q >= 2 * Q) return;
    int node = (q < Q) ? qs[q] : qd[q - Q];
    ((float4*)(out + (size_t)q * FD))[lane] = ((const float4*)(h + (size_t)node * FD))[lane];
}

extern "C" void kernel_launch(void* const* d_in, const int* in_sizes, int n_in,
                              void* d_out, int out_size, void* d_ws, size_t ws_size,
                              hipStream_t stream) {
    (void)in_sizes; (void)n_in; (void)out_size; (void)ws_size;
    const float* emb  = (const float*)d_in[0];
    const float* W1   = (const float*)d_in[1];
    const float* b1   = (const float*)d_in[2];
    const float* W2   = (const float*)d_in[3];
    const float* b2   = (const float*)d_in[4];
    const int*   esrc = (const int*)d_in[5];   // [NR][NE]
    const int*   edst = (const int*)d_in[6];   // [NR][NE]
    const int*   qsrc = (const int*)d_in[7];
    const int*   qdst = (const int*)d_in[8];
    float*       out  = (float*)d_out;

    // workspace carve-out (~131 MB), 256B aligned slices
    char*  base = (char*)d_ws;
    size_t off  = 0;
    auto alloc = [&](size_t bytes) -> void* {
        void* p = base + off;
        off = (off + bytes + 255) & ~(size_t)255;
        return p;
    };
    _Float16* Xh   = (_Float16*)alloc((size_t)NN * FD * sizeof(_Float16)); // X (then H1) in f16
    _Float16* Wt1  = (_Float16*)alloc((size_t)NR * FD * FD * sizeof(_Float16));
    _Float16* Wt2  = (_Float16*)alloc((size_t)NR * FD * FD * sizeof(_Float16));
    float*    ssrc = (float*)alloc((size_t)NR * NN * sizeof(float));
    float*    sdst = (float*)alloc((size_t)NR * NN * sizeof(float));
    float*    Z    = (float*)alloc((size_t)NN * FD * sizeof(float));       // per-rel GEMM out, reused as h2
    float*    agg  = (float*)alloc((size_t)NN * FD * sizeof(float));

    const int totalNF  = NN * FD;                    // 12.8M
    const int gemmGrid = (NN + 127) / 128;           // 8 waves x 16 rows per block

    // degrees -> rsqrt scale factors (exact integer-valued float sums: deterministic)
    (void)hipMemsetAsync(ssrc, 0, (size_t)NR * NN * sizeof(float), stream);
    (void)hipMemsetAsync(sdst, 0, (size_t)NR * NN * sizeof(float), stream);
    k_deg<<<(NR * NE + 255) / 256, 256, 0, stream>>>(esrc, edst, ssrc, sdst, NR * NE);
    k_rsqrt<<<(NR * NN + 255) / 256, 256, 0, stream>>>(ssrc, sdst, NR * NN);

    // precision prep: f16 activations + transposed f16 weights
    k_cast_half<<<(totalNF / 4 + 255) / 256, 256, 0, stream>>>(emb, Xh, totalNF / 4);
    k_prep_w<<<(NR * FD * FD + 255) / 256, 256, 0, stream>>>(W1, Wt1, NR * FD * FD);
    k_prep_w<<<(NR * FD * FD + 255) / 256, 256, 0, stream>>>(W2, Wt2, NR * FD * FD);

    // ---- layer 1 ----
    (void)hipMemsetAsync(agg, 0, (size_t)totalNF * sizeof(float), stream);
    for (int r = 0; r < NR; ++r) {
        k_gemm_wmma<<<gemmGrid, 256, 0, stream>>>(Xh, Wt1 + (size_t)r * FD * FD, Z, NN);
        k_edge<<<(NE * 32 + 255) / 256, 256, 0, stream>>>(Z, esrc + (size_t)r * NE,
                                                          edst + (size_t)r * NE,
                                                          ssrc + (size_t)r * NN,
                                                          sdst + (size_t)r * NN, agg, NE);
    }
    // h1 = tanh(agg + sum_r b1) written straight back as f16 GEMM input
    k_finalize<<<(totalNF + 255) / 256, 256, 0, stream>>>(agg, b1, nullptr, Xh, totalNF);

    // ---- layer 2 ----
    (void)hipMemsetAsync(agg, 0, (size_t)totalNF * sizeof(float), stream);
    for (int r = 0; r < NR; ++r) {
        k_gemm_wmma<<<gemmGrid, 256, 0, stream>>>(Xh, Wt2 + (size_t)r * FD * FD, Z, NN);
        k_edge<<<(NE * 32 + 255) / 256, 256, 0, stream>>>(Z, esrc + (size_t)r * NE,
                                                          edst + (size_t)r * NE,
                                                          ssrc + (size_t)r * NN,
                                                          sdst + (size_t)r * NN, agg, NE);
    }
    // h2 = tanh(agg + sum_r b2) -> reuse Z as f32 node output
    k_finalize<<<(totalNF + 255) / 256, 256, 0, stream>>>(agg, b2, Z, nullptr, totalNF);

    // final query gather (2*4096 rows of 128 f32)
    k_gather<<<(2 * NQ * 32 + 255) / 256, 256, 0, stream>>>(Z, qsrc, qdst, out, NQ);
}
